// tDPMPN_420906795221
// MI455X (gfx1250) — matrix operand order repaired and checked by reading, not compiled, not run on hardware
//
#include <hip/hip_runtime.h>
#include <hip/hip_bf16.h>
#include <stdint.h>

#define EDGES 131072
#define DIM   256
#define DSMF  64
#define NQ    128
#define NSEG  16384
#define NNEW  32768

typedef __bf16 bf16_t;
typedef bf16_t       v16bf __attribute__((ext_vector_type(16)));
typedef bf16_t       v2bf  __attribute__((ext_vector_type(2)));
typedef float        v16f  __attribute__((ext_vector_type(16)));
typedef float        v2f   __attribute__((ext_vector_type(2)));
typedef float        v8f   __attribute__((ext_vector_type(8)));
typedef unsigned int v8u   __attribute__((ext_vector_type(8)));

// ---- workspace layout (in 4-byte dwords) ----
#define WS_PKPROJ 0         // 4 nt * 8 kt * 32 lanes * 8 dw = 8192
#define WS_PKGL   8192      // 4*10*32*8 = 10240
#define WS_PKGR   18432     // 10240
#define WS_PKGC   28672     // 4*2*32*8 = 2048
#define WS_PKLIN  30720     // 16*8*32*8 = 32768
#define WS_QS     63488     // 128*64
#define WS_QR     71680
#define WS_QT     79872
#define WS_LOGIT  88064     // E
#define WS_Z      219136    // E
#define WS_SMAX   350208    // NSEG
#define WS_SSUM   366592    // NSEG  (end = 382976 dwords ~ 1.5 MB)

#define LDS_FENCE() asm volatile("s_wait_dscnt 0" ::: "memory")

// ---- bf16 helpers: let the backend use V_CVT_PK_BF16_F32 ----
__device__ __forceinline__ unsigned short f2bf(float x) {
  return __builtin_bit_cast(unsigned short, (bf16_t)x);
}
__device__ __forceinline__ unsigned int pk2bf(float a, float b) {
  v2f f; f[0] = a; f[1] = b;
  v2bf h = __builtin_convertvector(f, v2bf);
  return __builtin_bit_cast(unsigned int, h);
}

__device__ __forceinline__ v8f wmma_bf16(v16bf a, v16bf b, v8f c) {
  return __builtin_amdgcn_wmma_f32_16x16x32_bf16(false, a, false, b, (short)0, c, false, false);
}

// B fragment from pre-packed blob: contiguous 32B per lane per tile
__device__ __forceinline__ v16bf load_b_frag(const unsigned int* blob, int tileIdx, int lane) {
  const uint4* p = (const uint4*)(blob + ((size_t)tileIdx * 32 + lane) * 8);
  uint4 lo = p[0], hi = p[1];
  v8u u; u[0]=lo.x; u[1]=lo.y; u[2]=lo.z; u[3]=lo.w; u[4]=hi.x; u[5]=hi.y; u[6]=hi.z; u[7]=hi.w;
  return __builtin_bit_cast(v16bf, u);
}

// A fragment (16x32 bf16) from row-major fp32 in global: rowk = per-lane row ptr + ktile*32
// lanes 0-15: row=lane, K 0-7 & 16-23 ; lanes 16-31: row=lane-16, K 8-15 & 24-31
__device__ __forceinline__ v16bf load_a_frag_g(const float* rowk, int lane) {
  int kb = (lane >> 4) << 3;
  const float4* p0 = (const float4*)(rowk + kb);
  const float4* p1 = (const float4*)(rowk + kb + 16);
  float4 a0 = p0[0], a1 = p0[1], b0 = p1[0], b1 = p1[1];
  v16f f = {a0.x, a0.y, a0.z, a0.w, a1.x, a1.y, a1.z, a1.w,
            b0.x, b0.y, b0.z, b0.w, b1.x, b1.y, b1.z, b1.w};
  return __builtin_convertvector(f, v16bf);
}

// A fragment from bf16 LDS staging tile (row-major, stride 320)
__device__ __forceinline__ v16bf load_a_frag_lds(const unsigned short* st, int kcol, int lane) {
  int r = lane & 15; int kb = (lane >> 4) << 3;
  const uint4* p0 = (const uint4*)(st + r*320 + kcol + kb);
  const uint4* p1 = (const uint4*)(st + r*320 + kcol + kb + 16);
  uint4 a = p0[0], b = p1[0];
  v8u u; u[0]=a.x; u[1]=a.y; u[2]=a.z; u[3]=a.w; u[4]=b.x; u[5]=b.y; u[6]=b.z; u[7]=b.w;
  return __builtin_bit_cast(v16bf, u);
}

__device__ __forceinline__ void atomicMaxF(float* a, float v) {
  if (v >= 0.0f) atomicMax((int*)a, __float_as_int(v));
  else           atomicMin((unsigned int*)a, __float_as_uint(v));
}

// ============ kernel: pack weights into WMMA-B bf16 layout ============
__device__ __forceinline__ void packOne(const float* W, int K, int KT, unsigned int* out, int j) {
  int dw   = j & 7;
  int frag = j >> 3;
  int lane = frag & 31;
  int tile = frag >> 5;
  int kt = tile % KT, nt = tile / KT;
  int n = nt * 16 + (lane & 15);
  int k = kt * 32 + ((lane >> 4) << 3) + (dw & 3) * 2 + (dw >> 2) * 16;
  out[j] = pk2bf(W[(size_t)n * K + k], W[(size_t)n * K + k + 1]);
}

__global__ void k_pack(const float* pw, const float* glw, const float* grw,
                       const float* gcw, const float* lnw, unsigned int* wsu) {
  int j = blockIdx.x * blockDim.x + threadIdx.x;
  if (j >= 63488) return;
  if      (j < 8192)  packOne(pw,  256, 8,  wsu + WS_PKPROJ, j);
  else if (j < 18432) packOne(glw, 320, 10, wsu + WS_PKGL,   j - 8192);
  else if (j < 28672) packOne(grw, 320, 10, wsu + WS_PKGR,   j - 18432);
  else if (j < 30720) packOne(gcw, 64,  2,  wsu + WS_PKGC,   j - 28672);
  else                packOne(lnw, 256, 8,  wsu + WS_PKLIN,  j - 30720);
}

// ============ kernel: project queries (fp32, tiny) ============
__global__ void k_qproj(const float* qsrc, const float* qrel, const float* qtim,
                        const float* pw, const float* pb,
                        float* oqs, float* oqr, float* oqt) {
  int i = blockIdx.x * blockDim.x + threadIdx.x;
  if (i >= 3 * NQ * DSMF) return;
  int msel = i / (NQ * DSMF);
  int rem  = i % (NQ * DSMF);
  int b = rem >> 6, o = rem & 63;
  const float* x = ((msel == 0) ? qsrc : (msel == 1) ? qrel : qtim) + (size_t)b * DIM;
  const float* w = pw + (size_t)o * DIM;
  float s = pb[o];
  #pragma unroll 8
  for (int k = 0; k < DIM; ++k) s += x[k] * w[k];
  float* out = (msel == 0) ? oqs : (msel == 1) ? oqr : oqt;
  out[b * DSMF + o] = s;
}

// ============ kernel: init outputs + segment buffers ============
__global__ void k_init(float* out, unsigned int* wsu) {
  size_t i = (size_t)blockIdx.x * blockDim.x + threadIdx.x;
  if (i < (size_t)NNEW * (DIM + 1)) out[i] = 0.0f;
  if (i < NSEG) { wsu[WS_SMAX + i] = 0xFF7FFFFFu; wsu[WS_SSUM + i] = 0u; }
}

// ---- 16x(K=256)x64 projection GEMM: gather A rows, store bf16 result to staging ----
__device__ __forceinline__ void proj_gemm_store(const float* arowbase, const unsigned int* pk,
                                                const float* bias, unsigned short* st,
                                                int colbase, int lane) {
  int r = lane & 15, hi = lane >> 4;
  v8f z8 = {0,0,0,0,0,0,0,0};
  v8f acc[4] = {z8, z8, z8, z8};
  #pragma unroll
  for (int kt = 0; kt < 8; ++kt) {
    v16bf a = load_a_frag_g(arowbase + kt * 32, lane);
    #pragma unroll
    for (int nt = 0; nt < 4; ++nt)
      acc[nt] = wmma_bf16(a, load_b_frag(pk, nt * 8 + kt, lane), acc[nt]);
  }
  #pragma unroll
  for (int nt = 0; nt < 4; ++nt) {
    float b = bias[nt * 16 + r];
    #pragma unroll
    for (int g = 0; g < 8; ++g)
      st[(g + 8 * hi) * 320 + colbase + nt * 16 + r] = f2bf(acc[nt][g] + b);
  }
}

// ---- 16x(K=320)x64 GEMM over the full staging tile ----
__device__ __forceinline__ void gemm320(const unsigned short* st, const unsigned int* pk,
                                        v8f acc[4], int lane) {
  #pragma unroll
  for (int t = 0; t < 10; ++t) {
    v16bf a = load_a_frag_lds(st, t * 32, lane);
    #pragma unroll
    for (int nt = 0; nt < 4; ++nt)
      acc[nt] = wmma_bf16(a, load_b_frag(pk, nt * 10 + t, lane), acc[nt]);
  }
}

// ============ kernel: per-edge logits + segment max ============
__global__ __launch_bounds__(128) void k_logits(
    const float* __restrict__ memb, const float* __restrict__ rel,
    const int* __restrict__ eeg, const int* __restrict__ evi, const int* __restrict__ evj,
    const int* __restrict__ esm,
    const float* __restrict__ proj_b, const float* __restrict__ gl_b,
    const float* __restrict__ gr_b, const float* __restrict__ gc_b,
    const unsigned int* wsu, float* wsf) {
  __shared__ unsigned short stage[4][16 * 320];   // per-wave 16x320 bf16 = left_x/right_x layout
  int lane = threadIdx.x & 31;
  int wid  = threadIdx.x >> 5;
  int tile = blockIdx.x * 4 + wid;
  int base = tile * 16;
  unsigned short* st = stage[wid];
  int r = lane & 15, hi = lane >> 4;

  const unsigned int* pkproj = wsu + WS_PKPROJ;
  const unsigned int* pkgl   = wsu + WS_PKGL;
  const unsigned int* pkgr   = wsu + WS_PKGR;
  const unsigned int* pkgc   = wsu + WS_PKGC;
  const float* qs = wsf + WS_QS;
  const float* qr = wsf + WS_QR;
  const float* qt = wsf + WS_QT;
  float* logits = wsf + WS_LOGIT;
  float* smax   = wsf + WS_SMAX;

  // stage query blocks into cols 128..319 (bf16)
  for (int i = lane; i < 3072; i += 32) {
    int blk = i >> 10;
    int rem = i & 1023;
    int m = rem >> 6, n = rem & 63;
    int eg = eeg[base + m];
    const float* q = (blk == 0) ? qs : (blk == 1) ? qr : qt;
    st[m * 320 + 128 + blk * 64 + n] = f2bf(q[eg * DSMF + n]);
  }

  // rel_p -> cols 64..127 (contiguous rows of rel_emb)
  proj_gemm_store(rel + (size_t)(base + r) * DIM, pkproj, proj_b, st, 64, lane);
  // vi_p -> cols 0..63 (gathered rows)
  int vi = evi[base + r];
  proj_gemm_store(memb + (size_t)vi * DIM, pkproj, proj_b, st, 0, lane);
  LDS_FENCE();

  // left = lrelu(left_x @ gL^T + b)
  v8f z8 = {0,0,0,0,0,0,0,0};
  v8f accL[4] = {z8, z8, z8, z8};
  gemm320(st, pkgl, accL, lane);
  #pragma unroll
  for (int nt = 0; nt < 4; ++nt) {
    float b = gl_b[nt * 16 + r];
    #pragma unroll
    for (int g = 0; g < 8; ++g) {
      float v = accL[nt][g] + b;
      accL[nt][g] = (v > 0.0f) ? v : 0.01f * v;
    }
  }

  // vj_p -> cols 0..63 (overwrite hidden block)
  int vj = evj[base + r];
  proj_gemm_store(memb + (size_t)vj * DIM, pkproj, proj_b, st, 0, lane);
  LDS_FENCE();

  // right_pre = lrelu(right_x @ gR^T + b) -> store bf16 to cols 0..63
  v8f accR[4] = {z8, z8, z8, z8};
  gemm320(st, pkgr, accR, lane);
  #pragma unroll
  for (int nt = 0; nt < 4; ++nt) {
    float b = gr_b[nt * 16 + r];
    #pragma unroll
    for (int g = 0; g < 8; ++g) {
      float v = accR[nt][g] + b;
      v = (v > 0.0f) ? v : 0.01f * v;
      st[(g + 8 * hi) * 320 + nt * 16 + r] = f2bf(v);
    }
  }
  LDS_FENCE();

  // right = right_pre @ gC^T + b
  v8f accC[4] = {z8, z8, z8, z8};
  #pragma unroll
  for (int t = 0; t < 2; ++t) {
    v16bf a = load_a_frag_lds(st, t * 32, lane);
    #pragma unroll
    for (int nt = 0; nt < 4; ++nt)
      accC[nt] = wmma_bf16(a, load_b_frag(pkgc, nt * 2 + t, lane), accC[nt]);
  }

  // logits[m] = sum_n left*right ; cross-lane reduce within each 16-lane half
  float p[8];
  #pragma unroll
  for (int g = 0; g < 8; ++g) p[g] = 0.0f;
  #pragma unroll
  for (int nt = 0; nt < 4; ++nt) {
    float b = gc_b[nt * 16 + r];
    #pragma unroll
    for (int g = 0; g < 8; ++g) p[g] += accL[nt][g] * (accC[nt][g] + b);
  }
  #pragma unroll
  for (int mask = 1; mask <= 8; mask <<= 1) {
    #pragma unroll
    for (int g = 0; g < 8; ++g) p[g] += __shfl_xor(p[g], mask, 32);
  }
  if (r == 0) {
    #pragma unroll
    for (int g = 0; g < 8; ++g) {
      int e = base + g + 8 * hi;
      float L = p[g];
      logits[e] = L;
      atomicMaxF(smax + esm[e], L);
    }
  }
}

// ============ kernel: exp + segment sum ============
__global__ void k_expsum(const float* logits, const float* smax, const int* __restrict__ esm,
                         float* zb, float* ssum) {
  int e = blockIdx.x * blockDim.x + threadIdx.x;
  if (e >= EDGES) return;
  int sg = esm[e];
  float z = __expf(logits[e] - smax[sg]);
  zb[e] = z;
  atomicAdd(ssum + sg, z);
}

// ============ kernel: trans, attending, msg GEMM + scatter-add ============
__global__ __launch_bounds__(256) void k_final(
    const float* __restrict__ memb,
    const int* __restrict__ evj, const int* __restrict__ esm, const int* __restrict__ eag,
    const float* __restrict__ natt, const float* __restrict__ lin_b,
    const unsigned int* wsu, const float* wsf,
    float* out_att, float* out_upd) {
  int lane = threadIdx.x & 31;
  int wid  = threadIdx.x >> 5;
  int tile = blockIdx.x * 8 + wid;
  int base = tile * 16;
  int r = lane & 15, hi = lane >> 4;
  const unsigned int* pklin = wsu + WS_PKLIN;
  const float* zb   = wsf + WS_Z;
  const float* ssum = wsf + WS_SSUM;

  float tr[8]; int ag[8];
  #pragma unroll
  for (int g = 0; g < 8; ++g) {
    int e = base + g + 8 * hi;
    float z = zb[e];
    tr[g] = z / ssum[esm[e]];
    ag[g] = eag[e];
  }
  if (r == 0) {
    #pragma unroll
    for (int g = 0; g < 8; ++g) {
      int e = base + g + 8 * hi;
      atomicAdd(out_att + ag[g], tr[g] * natt[e]);
    }
  }

  // msg = hidden_vj_orig @ lin^T + b ; scaled by trans, scattered by agg_seg
  int vj = evj[base + r];
  const float* arow = memb + (size_t)vj * DIM;
  v16bf af[8];
  #pragma unroll
  for (int kt = 0; kt < 8; ++kt) af[kt] = load_a_frag_g(arow + kt * 32, lane);

  for (int nt = 0; nt < 16; ++nt) {
    v8f acc = {0,0,0,0,0,0,0,0};
    #pragma unroll
    for (int kt = 0; kt < 8; ++kt)
      acc = wmma_bf16(af[kt], load_b_frag(pklin, nt * 8 + kt, lane), acc);
    int n = nt * 16 + r;
    float lb = lin_b[n];
    #pragma unroll
    for (int g = 0; g < 8; ++g)
      atomicAdd(out_upd + (size_t)ag[g] * DIM + n, (acc[g] + lb) * tr[g]);
  }
}

// ============ launcher ============
extern "C" void kernel_launch(void* const* d_in, const int* in_sizes, int n_in,
                              void* d_out, int out_size, void* d_ws, size_t ws_size,
                              hipStream_t stream) {
  const float* natt = (const float*)d_in[0];
  const float* memb = (const float*)d_in[1];
  const float* rel  = (const float*)d_in[2];
  const float* qsrc = (const float*)d_in[3];
  const float* qrel = (const float*)d_in[4];
  const float* qtim = (const float*)d_in[5];
  const float* pw   = (const float*)d_in[6];
  const float* pb   = (const float*)d_in[7];
  const float* glw  = (const float*)d_in[8];
  const float* glb  = (const float*)d_in[9];
  const float* grw  = (const float*)d_in[10];
  const float* grb  = (const float*)d_in[11];
  const float* gcw  = (const float*)d_in[12];
  const float* gcb  = (const float*)d_in[13];
  const float* lnw  = (const float*)d_in[14];
  const float* lnb  = (const float*)d_in[15];
  const int* eeg = (const int*)d_in[16];
  const int* evi = (const int*)d_in[17];
  const int* evj = (const int*)d_in[18];
  const int* esm = (const int*)d_in[19];
  const int* eag = (const int*)d_in[20];

  unsigned int* wsu = (unsigned int*)d_ws;
  float*        wsf = (float*)d_ws;
  float* out_att = (float*)d_out;
  float* out_upd = out_att + NNEW;

  k_pack<<<248, 256, 0, stream>>>(pw, glw, grw, gcw, lnw, wsu);
  k_qproj<<<96, 256, 0, stream>>>(qsrc, qrel, qtim, pw, pb,
                                  wsf + WS_QS, wsf + WS_QR, wsf + WS_QT);
  k_init<<<(NNEW * (DIM + 1) + 255) / 256, 256, 0, stream>>>(out_att, wsu);
  k_logits<<<EDGES / 64, 128, 0, stream>>>(memb, rel, eeg, evi, evj, esm,
                                           pb, glb, grb, gcb, wsu, wsf);
  k_expsum<<<EDGES / 256, 256, 0, stream>>>(wsf + WS_LOGIT, wsf + WS_SMAX, esm,
                                            wsf + WS_Z, wsf + WS_SSUM);
  k_final<<<EDGES / 128, 256, 0, stream>>>(memb, evj, esm, eag, natt, lnb,
                                           wsu, wsf, out_att, out_upd);
}
